// DTHyperNet_81673098101278
// MI455X (gfx1250) — matrix operand order repaired and checked by reading, not compiled, not run on hardware
//
#include <hip/hip_runtime.h>
#include <hip/hip_bf16.h>

// ---------------------------------------------------------------------------
// DTHyperNet for MI455X (gfx1250), wave32 + v_wmma_f32_16x16x32_bf16.
// Weights pre-swizzled into WMMA-B fragment order: every B fragment is two
// global_load_b128 at an immediate offset from a loop-invariant per-lane base.
// A fragments are two ds_load_b128 from an LDS-staged activation tile.
// fi/fs/lnc GEMMs fused with softmax / routing / leaf-mix epilogues.
// ---------------------------------------------------------------------------

typedef __attribute__((ext_vector_type(16))) __bf16 v16bf;
typedef __attribute__((ext_vector_type(8)))  float  v8f;

#define EPSV 1e-5f

union FragB16 { uint4 q[2]; v16bf v; };

__device__ __forceinline__ float bf2f(unsigned short u) {
    return __uint_as_float(((unsigned)u) << 16);
}
__device__ __forceinline__ unsigned short f2bf(float f) {
    unsigned u = __float_as_uint(f);
    u += 0x7fffu + ((u >> 16) & 1u);      // round-to-nearest-even
    return (unsigned short)(u >> 16);
}
__device__ __forceinline__ v8f zero8() {
    v8f v = {0.f,0.f,0.f,0.f,0.f,0.f,0.f,0.f};
    return v;
}
__device__ __forceinline__ v16bf load_frag_g(const unsigned short* __restrict__ p) {
    FragB16 f;
    f.q[0] = *(const uint4*)p;
    f.q[1] = *(const uint4*)(p + 8);
    return f.v;
}
// A fragment (contiguous k order, matching the B packing): two contiguous 16B runs in a row-major
// LDS row: K = [half*16, half*16+8) and [half*16+8, half*16+16).
#define LOAD_AFRAG(dst, AhArr, rowBase, kbase, halfv) do {                 \
    FragB16 _f;                                                            \
    _f.q[0] = *(uint4*)&AhArr[(rowBase) + (kbase) + (halfv) * 16];         \
    _f.q[1] = *(uint4*)&AhArr[(rowBase) + (kbase) + 8 + (halfv) * 16];     \
    (dst) = _f.v; } while (0)

// ---------------------------------------------------------------- convert ---
__global__ void cvt_kernel(const float* __restrict__ in,
                           unsigned short* __restrict__ out, int n) {
    int i = blockIdx.x * blockDim.x + threadIdx.x;
    if (i < n) out[i] = f2bf(in[i]);
}

// W (512 x N, row-major f32) -> bf16 WMMA-B fragment layout.
// Fragment f = (t*16 + kb)*32 + half*16 + n holds 16 contiguous bf16:
//   elem j -> W[(kb*32 + half*16 + j)*N + t*16 + n]
__global__ void cvt_frag_kernel(const float* __restrict__ in,
                                unsigned short* __restrict__ out, int N) {
    int f = blockIdx.x * blockDim.x + threadIdx.x;   // N*32 fragments
    if (f >= N * 32) return;
    int n  = f & 15;
    int h  = (f >> 4) & 1;
    int kb = (f >> 5) & 15;
    int t  = f >> 9;
    int col   = t * 16 + n;
    int kbase = kb * 32 + h * 16;
    unsigned short* o = out + (size_t)f * 16;
    #pragma unroll
    for (int j = 0; j < 16; ++j)
        o[j] = f2bf(in[(size_t)(kbase + j) * N + col]);
}

// w_lc (512 x 1600) -> per-leaf zero-padded 512x112 fragment layout.
__global__ void cvt_lc_kernel(const float* __restrict__ in,
                              unsigned short* __restrict__ out) {
    int idx = blockIdx.x * blockDim.x + threadIdx.x;   // 16 * 3584
    if (idx >= 16 * 3584) return;
    int l = idx / 3584;
    int f = idx % 3584;
    int n  = f & 15;
    int h  = (f >> 4) & 1;
    int kb = (f >> 5) & 15;
    int t  = f >> 9;
    int col   = t * 16 + n;            // 0..111
    int kbase = kb * 32 + h * 16;
    unsigned short* o = out + (size_t)idx * 16;
    #pragma unroll
    for (int j = 0; j < 16; ++j)
        o[j] = (col < 100) ? f2bf(in[(size_t)(kbase + j) * 1600 + l * 100 + col])
                           : (unsigned short)0;
}

// ------------------------------------------------------ fused 512x512 GEMM ---
// O = act(bn(A @ W + bias)) [+ res].  Block tile 64x256; 8 waves as 2xM, 4xN;
// each wave computes 32x64 (8 wmma / k-step, B frags shared across 2 M rows).
// mode: 0 = relu-bn, 1 = gelu-bn, 2 = gelu-bn + residual.
__global__ __launch_bounds__(256)
void gemm512_kernel(const unsigned short* __restrict__ A,
                    const unsigned short* __restrict__ Wf,   // frag layout N=512
                    unsigned short* __restrict__ O,
                    const float* __restrict__ bias,
                    const float* __restrict__ gamma,
                    const float* __restrict__ beta,
                    const unsigned short* __restrict__ res,
                    int mode) {
    __shared__ alignas(16) unsigned short Ah[64 * 512];    // 64KB

    const int tid  = threadIdx.x;
    const int lane = tid & 31;
    const int wave = tid >> 5;
    const int wm   = wave & 1;        // 2 M sub-blocks of 32 rows
    const int wn   = wave >> 1;       // 4 N sub-blocks of 64 cols
    const int half = lane >> 4;
    const int lm   = lane & 15;
    const int m0   = blockIdx.y * 64;
    const int n0   = blockIdx.x * 256;

    for (int i = tid * 8; i < 64 * 512; i += 2048)
        *(uint4*)&Ah[i] = *(const uint4*)&A[(m0 + (i >> 9)) * 512 + (i & 511)];
    __syncthreads();

    v8f acc[2][4];
    #pragma unroll
    for (int ms = 0; ms < 2; ++ms)
        #pragma unroll
        for (int t = 0; t < 4; ++t) acc[ms][t] = zero8();

    const int rowBase0 = (wm * 32 + lm) * 512;
    // loop-invariant per-lane fragment base; all frag loads use imm offsets
    const unsigned short* bpBase =
        Wf + ((size_t)(((n0 >> 4) + wn * 4) * 512) + lane) * 16;

    #pragma unroll
    for (int kb = 0; kb < 16; ++kb) {
        v16bf a0, a1;
        LOAD_AFRAG(a0, Ah, rowBase0, kb * 32, half);
        LOAD_AFRAG(a1, Ah, rowBase0 + 16 * 512, kb * 32, half);
        #pragma unroll
        for (int t = 0; t < 4; ++t) {
            v16bf bv = load_frag_g(bpBase + t * 8192 + kb * 512);
            acc[0][t] = __builtin_amdgcn_wmma_f32_16x16x32_bf16(
                false, a0, false, bv, (short)0, acc[0][t], false, false);
            acc[1][t] = __builtin_amdgcn_wmma_f32_16x16x32_bf16(
                false, a1, false, bv, (short)0, acc[1][t], false, false);
        }
    }

    const float inv = rsqrtf(1.0f + EPSV);
    #pragma unroll
    for (int t = 0; t < 4; ++t) {
        int col = n0 + wn * 64 + t * 16 + lm;
        float b  = bias[col];
        float g  = gamma[col] * inv;
        float be = beta[col];
        #pragma unroll
        for (int ms = 0; ms < 2; ++ms) {
            #pragma unroll
            for (int j = 0; j < 8; ++j) {
                int row = m0 + wm * 32 + ms * 16 + half * 8 + j;
                float y = (acc[ms][t][j] + b) * g + be;
                if (mode == 0) {
                    y = fmaxf(y, 0.0f);
                } else {
                    y = 0.5f * y * (1.0f + erff(y * 0.70710678118654752f));
                    if (mode == 2) y += bf2f(res[row * 512 + col]);
                }
                O[row * 512 + col] = f2bf(y);
            }
        }
    }
}

// -------------------------------------------- fused fi/fs + softmax + sd ----
// Grid (B/16, 15): full 16x512 fi/fs tiles per node; fi -> LDS (exp cached
// in place), fs stays in registers; node_sd = sigmoid((Σe·x − Σe·fs)/Σe).
__global__ __launch_bounds__(256)
void node_kernel(const unsigned short* __restrict__ Hb,   // Bx512 bf16
                 const unsigned short* __restrict__ Wfi,  // frag layout N=7680
                 const unsigned short* __restrict__ Wfs,  // frag layout N=7680
                 const float* __restrict__ bfi,
                 const float* __restrict__ bfs,
                 const float* __restrict__ Xf,            // Bx512 f32
                 float* __restrict__ nodesd) {             // Bx15
    __shared__ alignas(16) unsigned short Ah[16 * 512];
    __shared__ float fiT[16 * 512];
    __shared__ float red[16 * 16];
    __shared__ float red2[16 * 16];
    __shared__ float mx[16], se[16], t1[16], t2[16];

    const int tid  = threadIdx.x;
    const int lane = tid & 31;
    const int wave = tid >> 5;
    const int half = lane >> 4;
    const int lm   = lane & 15;
    const int b0   = blockIdx.x * 16;
    const int node = blockIdx.y;
    const int nc0  = node * 512;

    for (int i = tid * 8; i < 16 * 512; i += 2048)
        *(uint4*)&Ah[i] = *(const uint4*)&Hb[b0 * 512 + i];
    __syncthreads();

    v8f afi[4], afs[4];
    #pragma unroll
    for (int t = 0; t < 4; ++t) { afi[t] = zero8(); afs[t] = zero8(); }

    const int rowBase = lm * 512;
    const size_t fragBase = ((size_t)((node * 32 + wave * 4) * 512) + lane) * 16;
    const unsigned short* bpFi = Wfi + fragBase;
    const unsigned short* bpFs = Wfs + fragBase;

    #pragma unroll
    for (int kb = 0; kb < 16; ++kb) {
        v16bf af;
        LOAD_AFRAG(af, Ah, rowBase, kb * 32, half);
        #pragma unroll
        for (int t = 0; t < 4; ++t) {
            afi[t] = __builtin_amdgcn_wmma_f32_16x16x32_bf16(
                false, af, false, load_frag_g(bpFi + t * 8192 + kb * 512),
                (short)0, afi[t], false, false);
            afs[t] = __builtin_amdgcn_wmma_f32_16x16x32_bf16(
                false, af, false, load_frag_g(bpFs + t * 8192 + kb * 512),
                (short)0, afs[t], false, false);
        }
    }

    // fi tile (+bias) to LDS
    #pragma unroll
    for (int t = 0; t < 4; ++t) {
        int col = wave * 64 + t * 16 + lm;
        float bb = bfi[nc0 + col];
        #pragma unroll
        for (int j = 0; j < 8; ++j)
            fiT[(half * 8 + j) * 512 + col] = afi[t][j] + bb;
    }
    __syncthreads();

    const int r = tid >> 4, s = tid & 15;
    {   // row-wise max
        float m = -3.4e38f;
        for (int f = s; f < 512; f += 16) m = fmaxf(m, fiT[r * 512 + f]);
        red[r * 16 + s] = m;
    }
    __syncthreads();
    if (tid < 16) {
        float m = -3.4e38f;
        for (int s2 = 0; s2 < 16; ++s2) m = fmaxf(m, red[tid * 16 + s2]);
        mx[tid] = m;
    }
    __syncthreads();
    {   // e = exp(fi - mx) cached in place; accumulate Σe and Σe·x together
        float es = 0.f, tx = 0.f, m = mx[r];
        for (int f = s; f < 512; f += 16) {
            float e = expf(fiT[r * 512 + f] - m);
            fiT[r * 512 + f] = e;
            es += e;
            tx += e * Xf[(b0 + r) * 512 + f];
        }
        red[r * 16 + s]  = es;
        red2[r * 16 + s] = tx;
    }
    __syncthreads();
    if (tid < 16) {
        float a = 0.f, b = 0.f;
        for (int s2 = 0; s2 < 16; ++s2) { a += red[tid * 16 + s2]; b += red2[tid * 16 + s2]; }
        se[tid] = a;
        t1[tid] = b;
        t2[tid] = 0.f;
    }
    __syncthreads();
    // Σe·fs straight from fs register accumulators (e cached in fiT)
    #pragma unroll
    for (int t = 0; t < 4; ++t) {
        int col = wave * 64 + t * 16 + lm;
        float bb = bfs[nc0 + col];
        #pragma unroll
        for (int j = 0; j < 8; ++j) {
            int rr = half * 8 + j;
            atomicAdd(&t2[rr], fiT[rr * 512 + col] * (afs[t][j] + bb));
        }
    }
    __syncthreads();
    if (tid < 16) {
        float arg = (t1[tid] - t2[tid]) / se[tid];
        nodesd[(b0 + tid) * 15 + node] = 1.0f / (1.0f + expf(-arg));
    }
}

// ------------------------------------------------------------- tree coeff ---
__global__ void coeff_kernel(const float* __restrict__ nodesd,
                             float* __restrict__ coeff, int total) {
    int i = blockIdx.x * blockDim.x + threadIdx.x;
    if (i >= total) return;
    int b = i >> 4, l = i & 15;
    const float* nd = nodesd + b * 15;
    float p1 = nd[0];
    float p2 = nd[1 + (l >> 3)];
    float p3 = nd[3 + (l >> 2)];
    float p4 = nd[7 + (l >> 1)];
    if ((l >> 3) & 1) p1 = 1.f - p1;
    if ((l >> 2) & 1) p2 = 1.f - p2;
    if ((l >> 1) & 1) p3 = 1.f - p3;
    if (l & 1)        p4 = 1.f - p4;
    coeff[i] = p1 * p2 * p3 * p4;
}

// ---------------------------------------- fused lnc GEMM + leaf mixing -----
// out[b,c] = Σ_l coeff[b,l]·(h[b,:]@w_lc[:,l*100+c] + b_lc[l*100+c]).
// Grid B/16, 8 waves; wave w owns cols [w*16, w*16+16); cols>=100 masked.
__global__ __launch_bounds__(256)
void out_kernel(const unsigned short* __restrict__ Hb,
                const unsigned short* __restrict__ Wlcf,  // per-leaf frag layout
                const float* __restrict__ blc,
                const float* __restrict__ coeff,          // Bx16
                float* __restrict__ out) {                 // Bx100
    __shared__ alignas(16) unsigned short Ah[16 * 512];
    __shared__ float cf[16 * 16];

    const int tid  = threadIdx.x;
    const int lane = tid & 31;
    const int wave = tid >> 5;
    const int half = lane >> 4;
    const int lm   = lane & 15;
    const int b0   = blockIdx.x * 16;

    for (int i = tid * 8; i < 16 * 512; i += 2048)
        *(uint4*)&Ah[i] = *(const uint4*)&Hb[b0 * 512 + i];
    cf[tid] = coeff[b0 * 16 + tid];
    __syncthreads();

    const int tw  = (wave < 7) ? wave : 6;   // wave 7 computes masked padding
    const int col = wave * 16 + lm;
    const int rowBase = lm * 512;
    float oacc[8] = {0.f,0.f,0.f,0.f,0.f,0.f,0.f,0.f};

    for (int l = 0; l < 16; ++l) {
        const unsigned short* bpL =
            Wlcf + ((size_t)(l * 3584 + tw * 512) + lane) * 16;
        v8f acc = zero8();
        #pragma unroll
        for (int kb = 0; kb < 16; ++kb) {
            v16bf af;
            LOAD_AFRAG(af, Ah, rowBase, kb * 32, half);
            acc = __builtin_amdgcn_wmma_f32_16x16x32_bf16(
                false, af, false, load_frag_g(bpL + kb * 512),
                (short)0, acc, false, false);
        }
        if (col < 100) {
            float bb = blc[l * 100 + col];
            #pragma unroll
            for (int j = 0; j < 8; ++j)
                oacc[j] += cf[(half * 8 + j) * 16 + l] * (acc[j] + bb);
        }
    }
    if (col < 100) {
        #pragma unroll
        for (int j = 0; j < 8; ++j)
            out[(b0 + half * 8 + j) * 100 + col] = oacc[j];
    }
}

// ---------------------------------------------------------------------------
extern "C" void kernel_launch(void* const* d_in, const int* in_sizes, int n_in,
                              void* d_out, int out_size, void* d_ws, size_t ws_size,
                              hipStream_t stream) {
    const float* x    = (const float*)d_in[0];
    const float* w_in = (const float*)d_in[1];
    const float* b_in = (const float*)d_in[2];
    const float* g0   = (const float*)d_in[3];
    const float* be0  = (const float*)d_in[4];
    const float* bw1  = (const float*)d_in[5];
    const float* bb1  = (const float*)d_in[6];
    const float* bg1  = (const float*)d_in[7];
    const float* bbe1 = (const float*)d_in[8];
    const float* bw2  = (const float*)d_in[9];
    const float* bb2  = (const float*)d_in[10];
    const float* bg2  = (const float*)d_in[11];
    const float* bbe2 = (const float*)d_in[12];
    const float* w_fi = (const float*)d_in[13];
    const float* b_fi = (const float*)d_in[14];
    const float* w_fs = (const float*)d_in[15];
    const float* b_fs = (const float*)d_in[16];
    const float* w_lc = (const float*)d_in[17];
    const float* b_lc = (const float*)d_in[18];
    float* out = (float*)d_out;

    char* base = (char*)d_ws;                       // ~46.4 MB used
    unsigned short* xb   = (unsigned short*)(base + 0);
    unsigned short* winb = (unsigned short*)(base + 8388608);
    unsigned short* bw1b = (unsigned short*)(base + 8912896);
    unsigned short* bw2b = (unsigned short*)(base + 9961472);
    unsigned short* wfib = (unsigned short*)(base + 11010048);
    unsigned short* wfsb = (unsigned short*)(base + 18874368);
    unsigned short* wlcb = (unsigned short*)(base + 26738688);  // 1,835,008 B
    unsigned short* hb   = (unsigned short*)(base + 28573696);
    unsigned short* ob   = (unsigned short*)(base + 36962304);
    float* nodesd        = (float*)(base + 45350912);
    float* coeff         = (float*)(base + 45842432);

    // x stays row-major bf16 (A side); all weights go to fragment layout.
    cvt_kernel<<<(8192 * 512 + 255) / 256, 256, 0, stream>>>(x, xb, 8192 * 512);
    cvt_frag_kernel<<<(512 * 32 + 255) / 256, 256, 0, stream>>>(w_in, winb, 512);
    for (int i = 0; i < 2; ++i) {
        cvt_frag_kernel<<<(512 * 32 + 255) / 256, 256, 0, stream>>>(
            bw1 + (size_t)i * 512 * 512, bw1b + (size_t)i * 512 * 512, 512);
        cvt_frag_kernel<<<(512 * 32 + 255) / 256, 256, 0, stream>>>(
            bw2 + (size_t)i * 512 * 512, bw2b + (size_t)i * 512 * 512, 512);
    }
    cvt_frag_kernel<<<(7680 * 32 + 255) / 256, 256, 0, stream>>>(w_fi, wfib, 7680);
    cvt_frag_kernel<<<(7680 * 32 + 255) / 256, 256, 0, stream>>>(w_fs, wfsb, 7680);
    cvt_lc_kernel<<<(16 * 3584 + 255) / 256, 256, 0, stream>>>(w_lc, wlcb);

    dim3 gg(2, 128), gb(256);
    // h = relu(bn(x @ w_in + b_in))
    gemm512_kernel<<<gg, gb, 0, stream>>>(xb, winb, hb, b_in, g0, be0, nullptr, 0);
    // residual blocks
    for (int i = 0; i < 2; ++i) {
        const unsigned short* w1 = bw1b + (size_t)i * 512 * 512;
        const unsigned short* w2 = bw2b + (size_t)i * 512 * 512;
        gemm512_kernel<<<gg, gb, 0, stream>>>(hb, w1, ob,
            bb1 + i * 512, bg1 + i * 512, bbe1 + i * 512, nullptr, 1);
        gemm512_kernel<<<gg, gb, 0, stream>>>(ob, w2, hb,
            bb2 + i * 512, bg2 + i * 512, bbe2 + i * 512, hb, 2);
    }
    // node routing probabilities (fused fi/fs GEMM + softmax + sigmoid)
    node_kernel<<<dim3(512, 15), gb, 0, stream>>>(hb, wfib, wfsb, b_fi, b_fs,
                                                  x, nodesd);
    // leaf coefficients
    coeff_kernel<<<512, 256, 0, stream>>>(nodesd, coeff, 8192 * 16);
    // fused lnc GEMM + coefficient mix
    out_kernel<<<512, gb, 0, stream>>>(hb, wlcb, b_lc, coeff, out);
}
